// CPCModel_84670985274154
// MI455X (gfx1250) — compile-verified, matrix operand order checked by reading, not run on hardware
//
#include <hip/hip_runtime.h>
#include <hip/hip_bf16.h>
#include <math.h>

// Problem constants (match reference)
enum { Bb = 32, Cc = 128, Tt = 256, Dd = 256, Hh = 512, Gg = 4 * Hh,
       SPc = 12, NFc = 64, Ll = Tt - SPc };  // Ll = 244

typedef __attribute__((ext_vector_type(8)))  _Float16 v8h;
typedef __attribute__((ext_vector_type(16))) _Float16 v16h;
typedef __attribute__((ext_vector_type(8)))  float    v8f;

// ---------------------------------------------------------------------------
// Fragment loader: 16x32 f16 A-fragment (or 32x16 B-fragment from an [N,K]
// row-major weight matrix). CDNA5 layout: lanes 0-15 hold row (lane&15),
// K offsets {0..7, 16..23}; lanes 16-31 hold K offsets {8..15, 24..31}.
// Two 16B loads per lane (global_load_b128 or ds_load_b128).
// ---------------------------------------------------------------------------
__device__ __forceinline__ v16h load_frag(const _Float16* tile, int ldk, int lane) {
  const _Float16* p = tile + (size_t)(lane & 15) * ldk + ((lane >> 4) << 3);
  v8h lo = *(const v8h*)(p);
  v8h hi = *(const v8h*)(p + 16);
  return __builtin_shufflevector(lo, hi, 0, 1, 2, 3, 4, 5, 6, 7,
                                 8, 9, 10, 11, 12, 13, 14, 15);
}

__device__ __forceinline__ v8f wmma_f16(v16h a, v16h b, v8f c) {
  return __builtin_amdgcn_wmma_f32_16x16x32_f16(false, a, false, b,
                                                (short)0, c, false, false);
}

__device__ __forceinline__ float sigf(float x) { return 1.0f / (1.0f + expf(-x)); }

// ---------------------------------------------------------------------------
// Prep kernels
// ---------------------------------------------------------------------------
__global__ void f32_to_f16_kernel(const float* __restrict__ in,
                                  _Float16* __restrict__ out, int n) {
  int i = blockIdx.x * blockDim.x + threadIdx.x;
  if (i < n) out[i] = (_Float16)in[i];
}

// in[R][Cn] (f32, row-major) -> out[Cn][R] (f16)
__global__ void transpose_f16_kernel(const float* __restrict__ in,
                                     _Float16* __restrict__ out, int R, int Cn) {
  int i = blockIdx.x * blockDim.x + threadIdx.x;
  if (i >= R * Cn) return;
  int r = i / Cn, c = i % Cn;
  out[(size_t)c * R + r] = (_Float16)in[(size_t)r * Cn + c];
}

// x[B,C,T] f32 -> xh[B,T,C] f16
__global__ void transpose_x_kernel(const float* __restrict__ x,
                                   _Float16* __restrict__ xh) {
  int i = blockIdx.x * blockDim.x + threadIdx.x;
  if (i >= Bb * Tt * Cc) return;
  int c = i % Cc;
  int t = (i / Cc) % Tt;
  int b = i / (Cc * Tt);
  xh[i] = (_Float16)x[((size_t)b * Cc + c) * Tt + t];
}

__global__ void vec_add_kernel(const float* __restrict__ a,
                               const float* __restrict__ b,
                               float* __restrict__ o, int n) {
  int i = blockIdx.x * blockDim.x + threadIdx.x;
  if (i < n) o[i] = a[i] + b[i];
}

// ---------------------------------------------------------------------------
// Generic NT GEMM: C[M,N] = A[M,K] (f16, row-major) x Bt[N,K]^T (f16) + bias
// K is a compile-time constant (128/256/512) so the k-loop fully unrolls into
// straight-line SSA: no register rotation moves, no tail branch, and the
// scheduler hoists loads ahead of their consuming WMMA (partial
// s_wait_loadcnt overlap). Block = 4 waves; each wave computes a 16x64 tile
// (1 A-frag reused across 4 B-frags). The block's 16xK A-strip is staged into
// LDS once via global_load_async_to_lds_b128 (ASYNCcnt path), so the 4 waves
// share one copy of A; B streams from L2. Grid: (N/256, M/16).
// ---------------------------------------------------------------------------
template <int K>
__global__ __launch_bounds__(128) void wmma_gemm_nt(
    const _Float16* __restrict__ A, const _Float16* __restrict__ Bt,
    const float* __restrict__ bias, float* __restrict__ Cf,
    _Float16* __restrict__ Ch, int M, int N) {
  __shared__ __align__(16) _Float16 sA[16 * K];

  const int lane  = threadIdx.x & 31;
  const int wave  = threadIdx.x >> 5;
  const int mbase = blockIdx.y * 16;
  const int nbase = blockIdx.x * 256 + wave * 64;

  // --- async-stage the 16xK A-strip (contiguous) into LDS ---
  {
    const char* src = (const char*)(A + (size_t)mbase * K);
    uint32_t lds0 = (uint32_t)(uintptr_t)(&sA[0]);
    constexpr int chunks = (16 * K * 2) / 16;  // 16B chunks
#pragma unroll
    for (int j = 0; j < chunks / 128; ++j) {
      int i = j * 128 + threadIdx.x;
      uint32_t loff = lds0 + (uint32_t)(i << 4);
      uint64_t g = (uint64_t)(uintptr_t)src + ((uint64_t)i << 4);
      asm volatile("global_load_async_to_lds_b128 %0, %1, off"
                   :: "v"(loff), "v"(g) : "memory");
    }
    asm volatile("s_wait_asynccnt 0x0" ::: "memory");
    __syncthreads();
  }

  v8f acc[4];
  v8f zero = {0.f, 0.f, 0.f, 0.f, 0.f, 0.f, 0.f, 0.f};
#pragma unroll
  for (int i = 0; i < 4; ++i) acc[i] = zero;

  const _Float16* Brow = Bt + (size_t)nbase * K;

#pragma unroll
  for (int kk = 0; kk < K; kk += 32) {
    v16h a = load_frag(sA + kk, K, lane);
#pragma unroll
    for (int tI = 0; tI < 4; ++tI) {
      v16h b = load_frag(Brow + (size_t)(tI * 16) * K + kk, K, lane);
      acc[tI] = wmma_f16(a, b, acc[tI]);
    }
  }

  const int mrow = (lane >> 4) * 8;
  const int ncol = lane & 15;
#pragma unroll
  for (int tI = 0; tI < 4; ++tI) {
    int n = nbase + tI * 16 + ncol;
    float bv = bias ? bias[n] : 0.f;
#pragma unroll
    for (int r = 0; r < 8; ++r) {
      int m = mbase + mrow + r;
      float v = acc[tI][r] + bv;
      Cf[(size_t)m * N + n] = v;
      if (Ch) Ch[(size_t)m * N + n] = (_Float16)v;
    }
  }
}

// ---------------------------------------------------------------------------
// LSTM recurrence (one layer). Batch-split: blockIdx.x in {0,1}, each block
// owns 16 batch rows (independent recurrences -> no inter-WG sync).
// 32 waves; wave w owns hidden units [w*16, w*16+16). Per timestep each wave
// runs 4 gate tiles (i,f,g,o) x 16 K-steps of v_wmma_f32_16x16x32_f16 against
// h held in LDS; c lives in registers; fused sigmoid/tanh epilogue.
// NOTE: the k-loop is kept ROLLED (partial unroll 2). A full unroll makes all
// 64 Whh B-fragments t-loop-invariant; LICM then hoists ~512 VGPRs of
// fragments and the allocator spills them to scratch (observed: inner loop
// fed by scratch_load_b128 reloads). A rolled loop keeps fragment addresses
// loop-variant, so they stay as global loads hitting L2 each timestep.
// xproj: [B,T,4H] f32 (already includes bih+bhh). hs out: [B,T,H] f16.
// Whh: [4H, H] f16 (NT layout).
// ---------------------------------------------------------------------------
__global__ __launch_bounds__(1024) void lstm_rec_kernel(
    const float* __restrict__ xproj, const _Float16* __restrict__ Whh,
    _Float16* __restrict__ hs) {
  __shared__ __align__(16) _Float16 hbuf[16 * Hh];  // 16 KB

  const int lane  = threadIdx.x & 31;
  const int wave  = threadIdx.x >> 5;
  const int nbase = wave * 16;
  const int bblk  = blockIdx.x * 16;

  for (int i = threadIdx.x; i < 16 * Hh; i += 1024) hbuf[i] = (_Float16)0.f;
  __syncthreads();

  float creg[8];
#pragma unroll
  for (int i = 0; i < 8; ++i) creg[i] = 0.f;

  v8f zero = {0.f, 0.f, 0.f, 0.f, 0.f, 0.f, 0.f, 0.f};
  const int ncol = nbase + (lane & 15);
  const int mrow = (lane >> 4) * 8;
  const _Float16* Wb = Whh + (size_t)nbase * Hh;

  for (int t = 0; t < Tt; ++t) {
    // Prefetch this timestep's input-projection gate values; these loads
    // drain while the recurrent GEMM below runs.
    float xv[4][8];
#pragma unroll
    for (int r = 0; r < 8; ++r) {
      int bg = bblk + mrow + r;
      size_t xb = ((size_t)bg * Tt + t) * Gg;
#pragma unroll
      for (int g = 0; g < 4; ++g) xv[g][r] = xproj[xb + g * Hh + ncol];
    }

    v8f acc[4];
#pragma unroll
    for (int g = 0; g < 4; ++g) acc[g] = zero;

#pragma unroll 2
    for (int kk = 0; kk < Hh; kk += 32) {
      v16h a = load_frag(&hbuf[kk], Hh, lane);
#pragma unroll
      for (int g = 0; g < 4; ++g) {
        v16h b = load_frag(Wb + (size_t)(g * Hh) * Hh + kk, Hh, lane);
        acc[g] = wmma_f16(a, b, acc[g]);
      }
    }
    __syncthreads();  // all reads of h_{t-1} complete

#pragma unroll
    for (int r = 0; r < 8; ++r) {
      int bl = mrow + r;          // local batch row 0..15
      int bg = bblk + bl;         // global batch
      float gi = acc[0][r] + xv[0][r];
      float gf = acc[1][r] + xv[1][r];
      float gg = acc[2][r] + xv[2][r];
      float go = acc[3][r] + xv[3][r];
      float c = sigf(gf) * creg[r] + sigf(gi) * tanhf(gg);
      float h = sigf(go) * tanhf(c);
      creg[r] = c;
      _Float16 hh = (_Float16)h;
      hbuf[(size_t)bl * Hh + ncol] = hh;
      hs[((size_t)bg * Tt + t) * Hh + ncol] = hh;
    }
    __syncthreads();  // h_t visible before next step's reads
  }
}

// ---------------------------------------------------------------------------
// InfoNCE: one wave per (l,b). Positive dot lane-parallel; 2 negatives per
// lane (64 total); wave-level logsumexp via __shfl_xor (wave32). Writes ce.
// ---------------------------------------------------------------------------
__global__ __launch_bounds__(256) void cpc_preds_kernel(
    const float* __restrict__ enc, const float* __restrict__ ctx,
    const int* __restrict__ idxs, float* __restrict__ ce) {
  int gw = (blockIdx.x * blockDim.x + threadIdx.x) >> 5;
  int lane = threadIdx.x & 31;
  if (gw >= Ll * Bb) return;
  int l = gw >> 5;   // / Bb
  int b = gw & 31;   // % Bb

  const float* cx = ctx + ((size_t)b * Tt + l) * Dd;
  const float* ps = enc + ((size_t)b * Tt + (l + SPc)) * Dd;

  float pp = 0.f;
  for (int d = lane; d < Dd; d += 32) pp += cx[d] * ps[d];
#pragma unroll
  for (int s = 16; s > 0; s >>= 1) pp += __shfl_xor(pp, s);

  float pv[2];
#pragma unroll
  for (int j = 0; j < 2; ++j) {
    int n = lane + j * 32;
    int idx = idxs[(size_t)l * (Bb * NFc) + b * NFc + n];
    int iseq = idx % (Tt - 1);
    int ib = idx / (Tt - 1);
    iseq += (iseq >= l + SPc) ? 1 : 0;
    const float* ng = enc + ((size_t)ib * Tt + iseq) * Dd;
    float s = 0.f;
    for (int d = 0; d < Dd; ++d) s += cx[d] * ng[d];
    pv[j] = s;
  }

  float m = fmaxf(pp, fmaxf(pv[0], pv[1]));
#pragma unroll
  for (int s = 16; s > 0; s >>= 1) m = fmaxf(m, __shfl_xor(m, s));
  float e = expf(pv[0] - m) + expf(pv[1] - m);
#pragma unroll
  for (int s = 16; s > 0; s >>= 1) e += __shfl_xor(e, s);
  float total = e + expf(pp - m);
  if (lane == 0) ce[gw] = (m + logf(total)) - pp;
}

// Deterministic single-block reduction: loss = sum(ce) / B
__global__ __launch_bounds__(256) void reduce_loss_kernel(
    const float* __restrict__ ce, float* __restrict__ out) {
  __shared__ float sm[256];
  float s = 0.f;
  for (int i = threadIdx.x; i < Ll * Bb; i += 256) s += ce[i];
  sm[threadIdx.x] = s;
  __syncthreads();
  if (threadIdx.x == 0) {
    float t = 0.f;
    for (int i = 0; i < 256; ++i) t += sm[i];
    out[0] = t / (float)Bb;
  }
}

// ---------------------------------------------------------------------------
// Host orchestration
// ---------------------------------------------------------------------------
extern "C" void kernel_launch(void* const* d_in, const int* in_sizes, int n_in,
                              void* d_out, int out_size, void* d_ws, size_t ws_size,
                              hipStream_t stream) {
  (void)in_sizes; (void)n_in; (void)out_size; (void)ws_size;
  const float* x    = (const float*)d_in[0];
  const float* Wenc = (const float*)d_in[1];
  const float* benc = (const float*)d_in[2];
  const float* Wih0 = (const float*)d_in[3];
  const float* Whh0 = (const float*)d_in[4];
  const float* bih0 = (const float*)d_in[5];
  const float* bhh0 = (const float*)d_in[6];
  const float* Wih1 = (const float*)d_in[7];
  const float* Whh1 = (const float*)d_in[8];
  const float* bih1 = (const float*)d_in[9];
  const float* bhh1 = (const float*)d_in[10];
  const float* Wp   = (const float*)d_in[11];
  const float* bp   = (const float*)d_in[12];
  const int*   idxs = (const int*)d_in[13];

  char* ws = (char*)d_ws;
  size_t off = 0;
  auto alloc = [&](size_t bytes) -> void* {
    void* p = ws + off;
    off += (bytes + 255) & ~(size_t)255;
    return p;
  };

  const int M = Bb * Tt;  // 8192
  _Float16* xh     = (_Float16*)alloc((size_t)M * Cc * 2);          // [B*T, C]
  _Float16* WencTh = (_Float16*)alloc((size_t)Dd * Cc * 2);         // [D, C]
  _Float16* Wih0h  = (_Float16*)alloc((size_t)Gg * Dd * 2);         // [4H, D]
  _Float16* Whh0h  = (_Float16*)alloc((size_t)Gg * Hh * 2);         // [4H, H]
  _Float16* Wih1h  = (_Float16*)alloc((size_t)Gg * Hh * 2);
  _Float16* Whh1h  = (_Float16*)alloc((size_t)Gg * Hh * 2);
  _Float16* WpTh   = (_Float16*)alloc((size_t)Dd * Hh * 2);         // [D, H]
  float*    b0     = (float*)alloc((size_t)Gg * 4);
  float*    b1     = (float*)alloc((size_t)Gg * 4);
  float*    encf   = (float*)alloc((size_t)M * Dd * 4);             // enc f32
  _Float16* ench   = (_Float16*)alloc((size_t)M * Dd * 2);          // enc f16
  float*    xproj  = (float*)alloc((size_t)M * Gg * 4);             // reused L0/L1
  _Float16* hs0    = (_Float16*)alloc((size_t)M * Hh * 2);
  _Float16* hs1    = (_Float16*)alloc((size_t)M * Hh * 2);
  float*    ctx    = (float*)alloc((size_t)M * Dd * 4);             // projection
  float*    cebuf  = (float*)alloc((size_t)Ll * Bb * 4);

  const int TPB = 256;
  auto g1 = [&](int n) { return dim3((n + TPB - 1) / TPB); };

  // --- weight prep ---
  f32_to_f16_kernel<<<g1(Gg * Dd), TPB, 0, stream>>>(Wih0, Wih0h, Gg * Dd);
  f32_to_f16_kernel<<<g1(Gg * Hh), TPB, 0, stream>>>(Whh0, Whh0h, Gg * Hh);
  f32_to_f16_kernel<<<g1(Gg * Hh), TPB, 0, stream>>>(Wih1, Wih1h, Gg * Hh);
  f32_to_f16_kernel<<<g1(Gg * Hh), TPB, 0, stream>>>(Whh1, Whh1h, Gg * Hh);
  transpose_f16_kernel<<<g1(Cc * Dd), TPB, 0, stream>>>(Wenc, WencTh, Cc, Dd);
  transpose_f16_kernel<<<g1(Hh * Dd), TPB, 0, stream>>>(Wp, WpTh, Hh, Dd);
  vec_add_kernel<<<g1(Gg), TPB, 0, stream>>>(bih0, bhh0, b0, Gg);
  vec_add_kernel<<<g1(Gg), TPB, 0, stream>>>(bih1, bhh1, b1, Gg);
  transpose_x_kernel<<<g1(Bb * Tt * Cc), TPB, 0, stream>>>(x, xh);

  dim3 blk(128);
  // --- encoder: enc = x^T @ Wenc + benc  (f32 + f16 copies), K = C = 128 ---
  wmma_gemm_nt<Cc><<<dim3(Dd / 256, M / 16), blk, 0, stream>>>(
      xh, WencTh, benc, encf, ench, M, Dd);
  // --- layer 0 input projection, K = D = 256 ---
  wmma_gemm_nt<Dd><<<dim3(Gg / 256, M / 16), blk, 0, stream>>>(
      ench, Wih0h, b0, xproj, (_Float16*)nullptr, M, Gg);
  // --- layer 0 recurrence ---
  lstm_rec_kernel<<<2, 1024, 0, stream>>>(xproj, Whh0h, hs0);
  // --- layer 1 input projection, K = H = 512 ---
  wmma_gemm_nt<Hh><<<dim3(Gg / 256, M / 16), blk, 0, stream>>>(
      hs0, Wih1h, b1, xproj, (_Float16*)nullptr, M, Gg);
  // --- layer 1 recurrence ---
  lstm_rec_kernel<<<2, 1024, 0, stream>>>(xproj, Whh1h, hs1);
  // --- projection: ctx = h1 @ Wp + bp, K = H = 512 ---
  wmma_gemm_nt<Hh><<<dim3(Dd / 256, M / 16), blk, 0, stream>>>(
      hs1, WpTh, bp, ctx, (_Float16*)nullptr, M, Dd);
  // --- loss ---
  cpc_preds_kernel<<<dim3((Ll * Bb * 32 + TPB - 1) / TPB), TPB, 0, stream>>>(
      encf, ctx, idxs, cebuf);
  reduce_loss_kernel<<<1, 256, 0, stream>>>(cebuf, (float*)d_out);
}
